// UnigramModel_46256797778391
// MI455X (gfx1250) — compile-verified, matrix operand order checked by reading, not run on hardware
//
#include <hip/hip_runtime.h>

// Problem sizes
#define VOCAB  50257
#define HID    512
#define LABELS 64
#define VTILE  128          // vocab rows per workgroup
#define VPAD   50304        // 393 * 128
#define NKC    16           // k-chunks of 32

typedef __attribute__((ext_vector_type(16))) __bf16 v16bf;
typedef __attribute__((ext_vector_type(8)))  float  v8f;
typedef __attribute__((ext_vector_type(4)))  unsigned u32x4;
typedef __attribute__((ext_vector_type(8)))  unsigned u32x8;

union BF16x16 { v16bf v; __bf16 e[16]; };
union F32x8   { v8f   v; float  e[8];  };

__device__ __forceinline__ void bf16_split(float x, __bf16& hi, __bf16& lo) {
  hi = (__bf16)x;
  lo = (__bf16)(x - (float)hi);
}

// Workspace layout: MT [VPAD x 64] f32 | B fragments (bf16 hi/lo) | cvec[64] f32
#define MT_FLOATS   ((size_t)VPAD * LABELS)
#define BFRAG_HALFS (NKC * 4 * 2 * 32 * 16)   // 65536 halves = 128 KB

__device__ __forceinline__ unsigned lds_off32(const void* p) {
  // Generic LDS address: low 32 bits are the LDS byte offset (ISA 10.2 aperture rules)
  return (unsigned)(size_t)p;
}

// ---- One-time: w2 -> pre-swizzled hi/lo bf16 WMMA B-fragments in d_ws ----
__global__ __launch_bounds__(128)
void prep_b_kernel(const float* __restrict__ w2, __bf16* __restrict__ bfrag)
{
  const int kc   = blockIdx.x;          // 0..15
  const int lt   = threadIdx.x >> 5;    // 0..3
  const int lane = threadIdx.x & 31;
  const int n    = lt * 16 + (lane & 15);
  const int kb1  = (lane < 16) ? 0 : 8;
  const float* bsrc = w2 + (size_t)n * HID + kc * 32;

  float tmp[16];
  *(float4*)&tmp[0]  = *(const float4*)(bsrc + kb1);
  *(float4*)&tmp[4]  = *(const float4*)(bsrc + kb1 + 4);
  *(float4*)&tmp[8]  = *(const float4*)(bsrc + kb1 + 16);
  *(float4*)&tmp[12] = *(const float4*)(bsrc + kb1 + 20);

  BF16x16 bh, bl;
  #pragma unroll
  for (int i = 0; i < 16; ++i) bf16_split(tmp[i], bh.e[i], bl.e[i]);

  v16bf* out = (v16bf*)bfrag;
  const int fid = ((kc * 4 + lt) * 2) * 32 + lane;  // hi plane; lo = fid + 32
  out[fid]      = bh.v;
  out[fid + 32] = bl.v;
}

// ---- One-time: cvec[l] = b2[l] + sum_h b1[h] * w2[l,h] ----
__global__ __launch_bounds__(64)
void cvec_kernel(const float* __restrict__ b1, const float* __restrict__ w2,
                 const float* __restrict__ b2, float* __restrict__ cvec)
{
  const int l = threadIdx.x;
  float s = b2[l];
  for (int h = 0; h < HID; ++h) s += b1[h] * w2[l * HID + h];
  cvec[l] = s;
}

// ---- Main GEMM: MT[v,l] = sum_h w1[h,v]*w2[l,h] + cvec[l] ----
#define AROWS   32
#define ASTRIDE 18   // padded row stride (floats): disjoint bank groups for half-waves

__global__ __launch_bounds__(256)
void logits_gemm_kernel(const float* __restrict__ w1,
                        const __bf16* __restrict__ bfrag,
                        const float* __restrict__ cvec,
                        float* __restrict__ mt)
{
  __shared__ float rawA[2][8][AROWS][ASTRIDE];   // 2 x 8 wave-private subtiles

  const int t    = threadIdx.x;
  const int wave = t >> 5;
  const int lane = t & 31;
  const int hi16 = lane >> 4;
  const int m    = lane & 15;
  const int vb   = blockIdx.x * VTILE;

  // Edge workgroup (tile crosses VOCAB): use clamped per-lane async b32 path.
  const bool edge = (vb + VTILE > VOCAB);

  int vg = vb + wave * 16 + m;
  if (vg > VOCAB - 1) vg = VOCAB - 1;

  v8f acc[4] = {};
  const v16bf* bf = (const v16bf*)bfrag;

  // -------- Path 1 (edge only): per-lane async b32 copies --------
  auto issueA = [&](int kc, int buf) {
    int      gbyte = ((kc * 32 + hi16) * VOCAB + vg) * 4;
    unsigned lbyte = lds_off32(&rawA[buf][wave][hi16][m]);
    #pragma unroll
    for (int i = 0; i < 16; ++i) {
      asm volatile("global_load_async_to_lds_b32 %0, %1, %2"
                   :: "v"(lbyte), "v"(gbyte), "s"(w1) : "memory");
      gbyte += 2 * VOCAB * 4;
      lbyte += 2 * ASTRIDE * 4;
    }
  };

  // -------- Path 2 (bulk): one TDM tensor_load_to_lds per wave per k-chunk ----
  // Wave-uniform scalars for the descriptor (SGPR-resident via readfirstlane).
  const unsigned waveu = (unsigned)__builtin_amdgcn_readfirstlane(wave);
  const unsigned ldsb0 = (unsigned)__builtin_amdgcn_readfirstlane(
                              lds_off32(&rawA[0][wave][0][0]));
  const unsigned ldsb1 = (unsigned)__builtin_amdgcn_readfirstlane(
                              lds_off32(&rawA[1][wave][0][0]));
  const unsigned long long gbase =
      (unsigned long long)(const void*)w1 + ((size_t)vb + (size_t)waveu * 16) * 4;

  auto issueTDM = [&](int kc, int buf) {
    const unsigned long long ga = gbase + (unsigned long long)kc * (32ull * VOCAB * 4ull);
    u32x4 g0;
    g0[0] = 1u;                                   // count=1, user descriptor
    g0[1] = buf ? ldsb1 : ldsb0;                  // lds_addr (bytes)
    g0[2] = (unsigned)ga;                         // global_addr[31:0]
    g0[3] = (unsigned)(ga >> 32) | (2u << 30);    // global_addr[56:32] | type=2
    u32x8 g1;
    g1[0] = (2u << 16)      // data_size = 4 bytes
          | (1u << 20)      // pad_enable
          | (3u << 22)      // pad_interval: every 16 DWORDs
          | (1u << 25);     // pad_amount: 2 DWORDs  -> row stride 18 floats
    g1[1] = ((unsigned)VOCAB & 0xFFFFu) << 16;    // tensor_dim0[15:0]
    g1[2] = ((unsigned)HID << 16) | ((unsigned)VOCAB >> 16); // tdim1[15:0] | tdim0[31:16]
    g1[3] = (16u << 16);                          // tile_dim0 = 16 | tdim1[31:16]=0
    g1[4] = 32u;                                  // tile_dim1 = 32, tile_dim2 = 0
    g1[5] = (unsigned)VOCAB;                      // tensor_dim0_stride[31:0]
    g1[6] = 0u;                                   // stride[47:32]=0, dim1_stride lo=0
    g1[7] = 0u;
    const u32x4 gz = {0u, 0u, 0u, 0u};            // groups 2/3: 2-D tensor, unused
    asm volatile("tensor_load_to_lds %0, %1, %2, %3"
                 :: "s"(g0), "s"(g1), "s"(gz), "s"(gz) : "memory");
  };

  if (edge) issueA(0, 0); else issueTDM(0, 0);

  #pragma unroll 1
  for (int kc = 0; kc < NKC; ++kc) {
    const int cur = kc & 1;
    if (kc + 1 < NKC) {
      // target buffer's previous-round DS reads must complete before overwrite
      asm volatile("s_wait_dscnt 0x0" ::: "memory");
      if (edge) {
        issueA(kc + 1, cur ^ 1);
        asm volatile("s_wait_asynccnt 0x10" ::: "memory");  // current chunk done
      } else {
        issueTDM(kc + 1, cur ^ 1);
        asm volatile("s_wait_tensorcnt 0x1" ::: "memory");  // in-order: cur done
      }
    } else {
      if (edge) asm volatile("s_wait_asynccnt 0x0" ::: "memory");
      else      asm volatile("s_wait_tensorcnt 0x0" ::: "memory");
    }

    // Gather + convert this wave's A fragments (hi/lo split, ~fp32 accuracy)
    BF16x16 ah, al;
    #pragma unroll
    for (int j = 0; j < 16; ++j) {
      const int kk = (hi16 == 0) ? ((j < 8) ? j : j + 8)
                                 : ((j < 8) ? j + 8 : j + 16);
      bf16_split(rawA[cur][wave][kk][m], ah.e[j], al.e[j]);
    }

    #pragma unroll
    for (int lt = 0; lt < 4; ++lt) {
      const int fid = ((kc * 4 + lt) * 2) * 32 + lane;
      const v16bf bh = bf[fid];        // prebuilt fragments, L2-hot b128 loads
      const v16bf bl = bf[fid + 32];
      acc[lt] = __builtin_amdgcn_wmma_f32_16x16x32_bf16(
                  false, ah.v, false, bh, (short)0, acc[lt], false, false);
      acc[lt] = __builtin_amdgcn_wmma_f32_16x16x32_bf16(
                  false, ah.v, false, bl, (short)0, acc[lt], false, false);
      acc[lt] = __builtin_amdgcn_wmma_f32_16x16x32_bf16(
                  false, al.v, false, bh, (short)0, acc[lt], false, false);
    }
  }

  // C/D layout: VGPR r -> (M = r + 8*hi16, N = lane&15)
  const int mrow = hi16 * 8;
  #pragma unroll
  for (int lt = 0; lt < 4; ++lt) {
    const int l  = lt * 16 + m;
    const float cl = cvec[l];
    F32x8 u; u.v = acc[lt];
    #pragma unroll
    for (int r = 0; r < 8; ++r) {
      const int v = vb + wave * 16 + mrow + r;
      mt[(size_t)v * LABELS + l] = u.e[r] + cl;
    }
  }
}

// ---- Step 2: out[t,:] = MT[tok_t,:] (256 B contiguous row per token) ----
__global__ __launch_bounds__(256)
void gather_kernel(const int* __restrict__ tok, const float* __restrict__ mt,
                   float* __restrict__ out, int ntok)
{
  const int gid   = blockIdx.x * 256 + threadIdx.x;
  const int token = gid >> 4;
  const int q     = gid & 15;
  if (token >= ntok) return;
  int id = tok[token];
  id = (id < 0) ? 0 : (id >= VOCAB ? VOCAB - 1 : id);
  const float4 val = ((const float4*)(mt + (size_t)id * LABELS))[q];
  ((float4*)(out + (size_t)token * LABELS))[q] = val;
}

// ---- Fallback if workspace too small ----
__global__ __launch_bounds__(64)
void naive_kernel(const int* __restrict__ tok, const float* __restrict__ w1,
                  const float* __restrict__ b1, const float* __restrict__ w2,
                  const float* __restrict__ b2, float* __restrict__ out, int ntok)
{
  const int token = blockIdx.x;
  const int l = threadIdx.x;
  if (token >= ntok) return;
  int id = tok[token];
  id = (id < 0) ? 0 : (id >= VOCAB ? VOCAB - 1 : id);
  float s = 0.0f;
  for (int h = 0; h < HID; ++h)
    s += (w1[(size_t)h * VOCAB + id] + b1[h]) * w2[l * HID + h];
  out[(size_t)token * LABELS + l] = s + b2[l];
}

extern "C" void kernel_launch(void* const* d_in, const int* in_sizes, int n_in,
                              void* d_out, int out_size, void* d_ws, size_t ws_size,
                              hipStream_t stream) {
  (void)n_in; (void)out_size;
  const int*   tok = (const int*)d_in[0];
  const float* w1  = (const float*)d_in[1];
  const float* b1  = (const float*)d_in[2];
  const float* w2  = (const float*)d_in[3];
  const float* b2  = (const float*)d_in[4];
  float* out = (float*)d_out;
  const int ntok = in_sizes[0];   // 32768

  const size_t mt_bytes    = MT_FLOATS * sizeof(float);
  const size_t bfrag_bytes = (size_t)BFRAG_HALFS * 2;
  const size_t need        = mt_bytes + bfrag_bytes + LABELS * sizeof(float);

  if (ws_size >= need) {
    float*  mt    = (float*)d_ws;
    __bf16* bfrag = (__bf16*)((char*)d_ws + mt_bytes);
    float*  cvec  = (float*)((char*)d_ws + mt_bytes + bfrag_bytes);

    prep_b_kernel<<<NKC, 128, 0, stream>>>(w2, bfrag);
    cvec_kernel<<<1, 64, 0, stream>>>(b1, w2, b2, cvec);
    logits_gemm_kernel<<<VPAD / VTILE, 256, 0, stream>>>(w1, bfrag, cvec, mt);
    const int tasks = ntok * 16;
    gather_kernel<<<(tasks + 255) / 256, 256, 0, stream>>>(tok, mt, out, ntok);
  } else {
    naive_kernel<<<ntok, 64, 0, stream>>>(tok, w1, b1, w2, b2, out, ntok);
  }
}
